// CausalSelfAttention_5755256176654
// MI455X (gfx1250) — matrix operand order repaired and checked by reading, not compiled, not run on hardware
//
#include <hip/hip_runtime.h>
#include <hip/hip_bf16.h>

// ---------------- types ----------------
typedef __attribute__((ext_vector_type(16))) _Float16 v16h;
typedef __attribute__((ext_vector_type(8)))  float    v8f;
typedef __attribute__((ext_vector_type(4)))  unsigned int u32x4;

#define B_DIM 16
#define T_DIM 256
#define C_DIM 6
#define E_DIM 512
#define H_DIM 8
#define HD    64
#define M_DIM (B_DIM*T_DIM*C_DIM)   // 24576
#define GRP   (B_DIM*H_DIM*C_DIM)   // 768
#define QKV_ELEMS (GRP*T_DIM*HD)    // 12,582,912 f16 elements per buffer

__device__ inline v8f zero8() {
    v8f r;
#pragma unroll
    for (int i = 0; i < 8; ++i) r[i] = 0.0f;
    return r;
}

__device__ inline v8f wmma_f16(v16h a, v16h b, v8f c) {
    return __builtin_amdgcn_wmma_f32_16x16x32_f16(false, a, false, b,
                                                  (short)0, c, false, false);
}

// ---- CDNA5 async global->LDS copy (ASYNCcnt-tracked, no VGPR round-trip) ----
__device__ inline void async_copy_b128(void* lds_dst, const void* gsrc) {
    unsigned int       l = (unsigned int)(unsigned long long)(uintptr_t)lds_dst;
    unsigned long long g = (unsigned long long)(uintptr_t)gsrc;
    asm volatile("global_load_async_to_lds_b128 %0, %1, off"
                 :: "v"(l), "v"(g) : "memory");
}
__device__ inline void wait_async0() {
    asm volatile("s_wait_asynccnt 0" ::: "memory");
}

// A fragment (16x32 f16, M x K): lane l holds row (l&15);
// lanes 0-15: K = 0..15, lanes 16-31: K = 16..31 (contiguous).
template <typename PT>
__device__ inline v16h load_frag_a(const PT* p) {
    union { u32x4 u[2]; v16h h; } t;
    t.u[0] = *(const u32x4*)(p);
    t.u[1] = *(const u32x4*)(p + 8);
    return t.h;
}
// B fragment (32x16 f16, K x N): lane l holds col (l&15);
// lanes 0-15: K = 0..15, lanes 16-31: K = 16..31 (contiguous).
template <typename PT>
__device__ inline v16h load_frag_b(const PT* p) {
    union { u32x4 u[2]; v16h h; } t;
    t.u[0] = *(const u32x4*)(p);
    t.u[1] = *(const u32x4*)(p + 8);
    return t.h;
}

// ---- GEMM staging helpers (rows of E_DIM f32 -> LDS f16 stride 72) ----
template <int NIT>
__device__ inline void load_tile_f32(const float* __restrict__ src, int row0,
                                     int k0, int tid, float4* v) {
#pragma unroll
    for (int it = 0; it < NIT; ++it) {
        int i = tid + it * 256;
        int r = i >> 4, c4 = (i & 15) << 2;
        v[it] = *(const float4*)(src + (size_t)(row0 + r) * E_DIM + k0 + c4);
    }
}
template <int NIT>
__device__ inline void cvt_store_tile(_Float16* base, const float4* v, int tid) {
#pragma unroll
    for (int it = 0; it < NIT; ++it) {
        int i = tid + it * 256;
        int r = i >> 4, c4 = (i & 15) << 2;
        _Float16* d = base + r * 72 + c4;
        d[0] = (_Float16)v[it].x; d[1] = (_Float16)v[it].y;
        d[2] = (_Float16)v[it].z; d[3] = (_Float16)v[it].w;
    }
}

// Shared WMMA micro-kernel: A tile row block (16) x 4 B column tiles.
// All 4 B fragments materialized before the WMMA chain -> one ds_load clause,
// amortized s_wait_dscnt, 4 independent accumulator chains.
__device__ inline void mma_16x64(const _Float16* xrow /* A row base +akb */,
                                 const _Float16* wbase /* B col base +bkb */,
                                 int wstride, v8f* acc) {
    v16h a  = load_frag_a(xrow);
    v16h b0 = load_frag_b(wbase + 0 * 16 * wstride);
    v16h b1 = load_frag_b(wbase + 1 * 16 * wstride);
    v16h b2 = load_frag_b(wbase + 2 * 16 * wstride);
    v16h b3 = load_frag_b(wbase + 3 * 16 * wstride);
    acc[0] = wmma_f16(a, b0, acc[0]);
    acc[1] = wmma_f16(a, b1, acc[1]);
    acc[2] = wmma_f16(a, b2, acc[2]);
    acc[3] = wmma_f16(a, b3, acc[3]);
}

// =====================================================================
// Kernel 1: fused QKV projection.  y = x @ W^T + b  (W is [N=512][K=512])
// Block tile 64(M) x 128(N), double-buffered LDS, software-pipelined loads.
// =====================================================================
__global__ void __launch_bounds__(256)
qkv_proj_kernel(const float* __restrict__ x,
                const float* __restrict__ Wq, const float* __restrict__ bq,
                const float* __restrict__ Wk, const float* __restrict__ bk,
                const float* __restrict__ Wv, const float* __restrict__ bv,
                _Float16* __restrict__ qo, _Float16* __restrict__ ko,
                _Float16* __restrict__ vto)
{
    __shared__ __align__(16) _Float16 Xs[2][64 * 72];
    __shared__ __align__(16) _Float16 Ws[2][128 * 72];

    const int which = blockIdx.z;
    const float* __restrict__ W    = (which == 0) ? Wq : (which == 1) ? Wk : Wv;
    const float* __restrict__ bias = (which == 0) ? bq : (which == 1) ? bk : bv;
    _Float16* __restrict__ dst     = (which == 0) ? qo : (which == 1) ? ko : vto;

    const int m0   = blockIdx.x * 64;
    const int n0   = blockIdx.y * 128;
    const int tid  = threadIdx.x;
    const int wave = tid >> 5, lane = tid & 31;
    const int wm = wave >> 1, wn = wave & 1;           // 4 x 2 wave grid
    const int l15 = lane & 15;
    const int hl  = lane >> 4;
    const int akb = hl ? 16 : 0;
    const int bkb = hl ? 16 : 0;

    v8f acc[4];
#pragma unroll
    for (int nt = 0; nt < 4; ++nt) acc[nt] = zero8();

    float4 vx[4], vw[8];
    load_tile_f32<4>(x, m0, 0, tid, vx);
    load_tile_f32<8>(W, n0, 0, tid, vw);
    cvt_store_tile<4>(Xs[0], vx, tid);
    cvt_store_tile<8>(Ws[0], vw, tid);
    __syncthreads();

    for (int kt = 0; kt < 8; ++kt) {
        const int cur = kt & 1, nxt = cur ^ 1;
        if (kt < 7) {                                  // issue next-tile loads early
            load_tile_f32<4>(x, m0, (kt + 1) * 64, tid, vx);
            load_tile_f32<8>(W, n0, (kt + 1) * 64, tid, vw);
        }
#pragma unroll
        for (int ks = 0; ks < 2; ++ks) {
            mma_16x64(&Xs[cur][(wm * 16 + l15) * 72 + ks * 32 + akb],
                      &Ws[cur][(wn * 64 + l15) * 72 + ks * 32 + bkb],
                      72, acc);
        }
        if (kt < 7) {
            cvt_store_tile<4>(Xs[nxt], vx, tid);
            cvt_store_tile<8>(Ws[nxt], vw, tid);
        }
        __syncthreads();
    }

    // epilogue: scatter into q/k/vT layouts (bias hoisted out of row loop)
    float bsv[4];
    int hdx[4], ddx[4];
#pragma unroll
    for (int nt = 0; nt < 4; ++nt) {
        int nc = n0 + wn * 64 + nt * 16 + l15;
        bsv[nt] = bias[nc];
        hdx[nt] = nc >> 6;
        ddx[nt] = nc & 63;
    }
#pragma unroll
    for (int j = 0; j < 8; ++j) {
        int m  = m0 + wm * 16 + j + hl * 8;
        int bb = m / (T_DIM * C_DIM);
        int rr = m % (T_DIM * C_DIM);
        int t  = rr / C_DIM, c = rr % C_DIM;
#pragma unroll
        for (int nt = 0; nt < 4; ++nt) {
            int gg = (bb * H_DIM + hdx[nt]) * C_DIM + c;
            size_t idx = (which == 2)
                       ? ((size_t)gg * HD + ddx[nt]) * T_DIM + t  // vT: [g][d][t]
                       : ((size_t)gg * T_DIM + t) * HD + ddx[nt]; // q,k: [g][t][d]
            dst[idx] = (_Float16)(acc[nt][j] + bsv[nt]);
        }
    }
}

// =====================================================================
// Kernel 2: fused causal attention per (b,h,c) group, flash-style.
// Block = 256 thr (8 waves); block handles 128 query rows; wave owns 16.
// K/V chunks staged via CDNA5 async global->LDS DMA.
// =====================================================================
__global__ void __launch_bounds__(256)
attn_kernel(const _Float16* __restrict__ qg, const _Float16* __restrict__ kg,
            const _Float16* __restrict__ vtg, _Float16* __restrict__ ybuf)
{
    __shared__ __align__(16) _Float16 Ks[128 * 64];    // [key][d]
    __shared__ __align__(16) _Float16 Vs[64 * 128];    // [d][key]
    __shared__ __align__(16) _Float16 Ps[8 * 16 * 32]; // per-wave P tile

    const int g    = blockIdx.x;
    const int half = blockIdx.y;
    const int r0wg = half * 128;
    const int tid  = threadIdx.x;
    const int wave = tid >> 5, lane = tid & 31;
    const int r0   = r0wg + wave * 16;
    const int l15  = lane & 15;
    const int hl   = lane >> 4;
    const int akb  = hl ? 16 : 0;
    const int bkb  = hl ? 16 : 0;
    const int rowoff = hl * 8;

    const size_t gbase = (size_t)g * (T_DIM * HD);
    const _Float16* __restrict__ qb  = qg  + gbase;
    const _Float16* __restrict__ kb_ = kg  + gbase;
    const _Float16* __restrict__ vtb = vtg + gbase;

    v16h aq[2];
#pragma unroll
    for (int ks = 0; ks < 2; ++ks)
        aq[ks] = load_frag_a(qb + (r0 + l15) * HD + ks * 32 + akb);

    v8f yacc[4];
#pragma unroll
    for (int nt = 0; nt < 4; ++nt) yacc[nt] = zero8();
    float rmax[8], rsum[8];
#pragma unroll
    for (int j = 0; j < 8; ++j) { rmax[j] = -3.0e38f; rsum[j] = 0.0f; }

    const float scale = 0.125f;                        // 1/sqrt(64)
    const int nchunk = half + 1;                       // causal: skip dead chunk

    for (int ch = 0; ch < nchunk; ++ch) {
        const int key0 = ch * 128;
        // ---- async-DMA stage K chunk [128][64] and V chunk [64][128] ----
#pragma unroll
        for (int it = 0; it < 4; ++it) {
            int i = tid + it * 256;                    // [0,1024)
            int kr = i >> 3, c8 = (i & 7) * 8;
            async_copy_b128(&Ks[kr * 64 + c8],
                            kb_ + (size_t)(key0 + kr) * HD + c8);
            int dr = i >> 4, e8 = (i & 15) * 8;
            async_copy_b128(&Vs[dr * 128 + e8],
                            vtb + (size_t)dr * T_DIM + key0 + e8);
        }
        wait_async0();
        __syncthreads();

        const int ltmax_raw = (r0 + 15 - key0) >> 4;
        const int ltmax = (ltmax_raw > 7) ? 7 : ltmax_raw;   // wave-uniform

        // ---- S = q @ k^T : process tile pairs (2 indep accumulator chains) ----
        v8f sacc[8];
#pragma unroll
        for (int lt = 0; lt < 8; ++lt) sacc[lt] = zero8();
#pragma unroll
        for (int ltp = 0; ltp < 4; ++ltp) {
            const int lt0 = 2 * ltp, lt1 = lt0 + 1;
            if (lt0 <= ltmax) {
                const bool do1 = (lt1 <= ltmax);
                v16h b00 = load_frag_b(&Ks[(lt0 * 16 + l15) * 64 + 0  + bkb]);
                v16h b01 = load_frag_b(&Ks[(lt0 * 16 + l15) * 64 + 32 + bkb]);
                v16h b10, b11;
                if (do1) {
                    b10 = load_frag_b(&Ks[(lt1 * 16 + l15) * 64 + 0  + bkb]);
                    b11 = load_frag_b(&Ks[(lt1 * 16 + l15) * 64 + 32 + bkb]);
                }
                sacc[lt0] = wmma_f16(aq[0], b00, sacc[lt0]);
                if (do1) sacc[lt1] = wmma_f16(aq[0], b10, sacc[lt1]);
                sacc[lt0] = wmma_f16(aq[1], b01, sacc[lt0]);
                if (do1) sacc[lt1] = wmma_f16(aq[1], b11, sacc[lt1]);
            }
        }

        // ---- scale + causal mask + chunk row-max ----
        float cmax[8];
#pragma unroll
        for (int j = 0; j < 8; ++j) cmax[j] = -3.0e38f;
#pragma unroll
        for (int lt = 0; lt < 8; ++lt) {
            if (lt <= ltmax) {
                int col = key0 + lt * 16 + l15;
#pragma unroll
                for (int j = 0; j < 8; ++j) {
                    int row = r0 + j + rowoff;
                    float s = sacc[lt][j] * scale;
                    s = (col <= row) ? s : -3.0e38f;
                    sacc[lt][j] = s;
                    cmax[j] = fmaxf(cmax[j], s);
                }
            }
        }
#pragma unroll
        for (int j = 0; j < 8; ++j) {                  // 16-lane half reduction
            float v = cmax[j];
            v = fmaxf(v, __shfl_xor(v, 1));
            v = fmaxf(v, __shfl_xor(v, 2));
            v = fmaxf(v, __shfl_xor(v, 4));
            v = fmaxf(v, __shfl_xor(v, 8));
            cmax[j] = v;
        }

        // ---- online softmax rescale ----
        float corr[8];
#pragma unroll
        for (int j = 0; j < 8; ++j) {
            float nm = fmaxf(rmax[j], cmax[j]);
            corr[j]  = __expf(rmax[j] - nm);
            rmax[j]  = nm;
            rsum[j] *= corr[j];
        }
#pragma unroll
        for (int nt = 0; nt < 4; ++nt)
#pragma unroll
            for (int j = 0; j < 8; ++j) yacc[nt][j] *= corr[j];

        // ---- P = exp(s - m), row sums ----
        float psum[8];
#pragma unroll
        for (int j = 0; j < 8; ++j) psum[j] = 0.0f;
#pragma unroll
        for (int lt = 0; lt < 8; ++lt) {
            if (lt <= ltmax) {
#pragma unroll
                for (int j = 0; j < 8; ++j) {
                    float p = __expf(sacc[lt][j] - rmax[j]);
                    sacc[lt][j] = p;
                    psum[j] += p;
                }
            }
        }
#pragma unroll
        for (int j = 0; j < 8; ++j) {
            float v = psum[j];
            v += __shfl_xor(v, 1);
            v += __shfl_xor(v, 2);
            v += __shfl_xor(v, 4);
            v += __shfl_xor(v, 8);
            rsum[j] += v;
        }

        // ---- Y += P @ V  (D-layout -> A-layout via per-wave LDS) ----
        _Float16* Pw = &Ps[wave * 16 * 32];
        const int submax = ltmax >> 1;
#pragma unroll
        for (int sub = 0; sub < 4; ++sub) {
            if (sub <= submax) {
                // preload V fragments BEFORE the P round-trip: the big
                // s_wait_dscnt 0 below covers them at no extra cost.
                v16h bv0 = load_frag_b(&Vs[(0 * 16 + l15) * 128 + sub * 32 + bkb]);
                v16h bv1 = load_frag_b(&Vs[(1 * 16 + l15) * 128 + sub * 32 + bkb]);
                v16h bv2 = load_frag_b(&Vs[(2 * 16 + l15) * 128 + sub * 32 + bkb]);
                v16h bv3 = load_frag_b(&Vs[(3 * 16 + l15) * 128 + sub * 32 + bkb]);
#pragma unroll
                for (int h2 = 0; h2 < 2; ++h2) {
                    const int lt = sub * 2 + h2;
#pragma unroll
                    for (int j = 0; j < 8; ++j) {
                        float p = (lt <= ltmax) ? sacc[lt][j] : 0.0f;
                        Pw[(j + rowoff) * 32 + h2 * 16 + l15] = (_Float16)p;
                    }
                }
                asm volatile("s_wait_dscnt 0" ::: "memory");
                v16h ap = load_frag_a(Pw + l15 * 32 + akb);
                yacc[0] = wmma_f16(ap, bv0, yacc[0]);
                yacc[1] = wmma_f16(ap, bv1, yacc[1]);
                yacc[2] = wmma_f16(ap, bv2, yacc[2]);
                yacc[3] = wmma_f16(ap, bv3, yacc[3]);
                // no trailing wait: LDS ops are in-order within a wave
            }
        }
        __syncthreads();
    }

    // ---- normalize, store to scrambled (B,T,H,C,hd) layout ----
    const int bb = g / (H_DIM * C_DIM);
    const int rem = g % (H_DIM * C_DIM);
    const int hh = rem / C_DIM, cc = rem % C_DIM;
#pragma unroll
    for (int j = 0; j < 8; ++j) {
        float inv = 1.0f / rsum[j];
        int t = r0 + j + rowoff;
        size_t rowbase = ((size_t)(bb * T_DIM + t)) * (C_DIM * E_DIM)
                       + (size_t)hh * (C_DIM * HD) + (size_t)cc * HD;
#pragma unroll
        for (int nt = 0; nt < 4; ++nt) {
            int d = nt * 16 + l15;
            ybuf[rowbase + d] = (_Float16)(yacc[nt][j] * inv);
        }
    }
}

// =====================================================================
// Kernel 3: output projection.  out = ybuf(f16, 24576x512) @ Wp^T + bp
// Block tile 64x128, double-buffered LDS; X tile streamed by async DMA,
// W tile register-pipelined f32->f16.
// =====================================================================
__global__ void __launch_bounds__(256)
out_proj_kernel(const _Float16* __restrict__ yin,
                const float* __restrict__ Wp, const float* __restrict__ bp,
                float* __restrict__ out)
{
    __shared__ __align__(16) _Float16 Xs[2][64 * 72];
    __shared__ __align__(16) _Float16 Ws[2][128 * 72];

    const int m0 = blockIdx.x * 64;
    const int n0 = blockIdx.y * 128;
    const int tid = threadIdx.x;
    const int wave = tid >> 5, lane = tid & 31;
    const int wm = wave >> 1, wn = wave & 1;
    const int l15 = lane & 15;
    const int hl = lane >> 4;
    const int akb = hl ? 16 : 0;
    const int bkb = hl ? 16 : 0;

    v8f acc[4];
#pragma unroll
    for (int nt = 0; nt < 4; ++nt) acc[nt] = zero8();

    auto stage_x = [&](int buf, int k0) {
#pragma unroll
        for (int it = 0; it < 2; ++it) {
            int i = tid + it * 256;
            int r = i >> 3, c8 = (i & 7) * 8;
            async_copy_b128(&Xs[buf][r * 72 + c8],
                            yin + (size_t)(m0 + r) * E_DIM + k0 + c8);
        }
    };

    float4 vw[8];
    stage_x(0, 0);
    load_tile_f32<8>(Wp, n0, 0, tid, vw);
    cvt_store_tile<8>(Ws[0], vw, tid);
    wait_async0();
    __syncthreads();

    for (int kt = 0; kt < 8; ++kt) {
        const int cur = kt & 1, nxt = cur ^ 1;
        if (kt < 7) {                                  // overlap DMA + loads w/ WMMA
            stage_x(nxt, (kt + 1) * 64);
            load_tile_f32<8>(Wp, n0, (kt + 1) * 64, tid, vw);
        }
#pragma unroll
        for (int ks = 0; ks < 2; ++ks) {
            mma_16x64(&Xs[cur][(wm * 16 + l15) * 72 + ks * 32 + akb],
                      &Ws[cur][(wn * 64 + l15) * 72 + ks * 32 + bkb],
                      72, acc);
        }
        if (kt < 7) {
            cvt_store_tile<8>(Ws[nxt], vw, tid);
            wait_async0();
        }
        __syncthreads();
    }

    float bsv[4];
    int ncv[4];
#pragma unroll
    for (int nt = 0; nt < 4; ++nt) {
        ncv[nt] = n0 + wn * 64 + nt * 16 + l15;
        bsv[nt] = bp[ncv[nt]];
    }
#pragma unroll
    for (int j = 0; j < 8; ++j) {
        int m = m0 + wm * 16 + j + hl * 8;
#pragma unroll
        for (int nt = 0; nt < 4; ++nt)
            out[(size_t)m * E_DIM + ncv[nt]] = acc[nt][j] + bsv[nt];
    }
}

// =====================================================================
extern "C" void kernel_launch(void* const* d_in, const int* in_sizes, int n_in,
                              void* d_out, int out_size, void* d_ws, size_t ws_size,
                              hipStream_t stream)
{
    const float* x  = (const float*)d_in[0];
    const float* Wq = (const float*)d_in[1];
    const float* bq = (const float*)d_in[2];
    const float* Wk = (const float*)d_in[3];
    const float* bk = (const float*)d_in[4];
    const float* Wv = (const float*)d_in[5];
    const float* bv = (const float*)d_in[6];
    const float* Wp = (const float*)d_in[7];
    const float* bp = (const float*)d_in[8];
    float* out = (float*)d_out;

    _Float16* ws   = (_Float16*)d_ws;
    _Float16* qbuf = ws;
    _Float16* kbuf = ws + (size_t)QKV_ELEMS;
    _Float16* vtbf = ws + (size_t)2 * QKV_ELEMS;
    _Float16* ybuf = ws + (size_t)3 * QKV_ELEMS;

    qkv_proj_kernel<<<dim3(M_DIM / 64, E_DIM / 128, 3), 256, 0, stream>>>(
        x, Wq, bq, Wk, bk, Wv, bv, qbuf, kbuf, vtbf);

    attn_kernel<<<dim3(GRP, 2), 256, 0, stream>>>(qbuf, kbuf, vtbf, ybuf);

    out_proj_kernel<<<dim3(M_DIM / 64, E_DIM / 128), 256, 0, stream>>>(
        ybuf, Wp, bp, out);
}
